// CrossAxialAttention_86981677678766
// MI455X (gfx1250) — matrix operand order repaired and checked by reading, not compiled, not run on hardware
//
#include <hip/hip_runtime.h>

#define DIM 64
#define HEADS 8
#define DKQ 8
#define IN_C 256
#define IN_SKIP 256
#define DV 32
#define EPS 1e-5f

typedef __attribute__((ext_vector_type(16))) __bf16 v16bf;
typedef __attribute__((ext_vector_type(8)))  float v8f;
typedef __attribute__((ext_vector_type(4)))  unsigned v4u;
typedef __attribute__((ext_vector_type(8)))  int v8i;
typedef __attribute__((ext_vector_type(4)))  int v4i;

union BFrag {
  v16bf v;
  uint4 q[2];
  unsigned short s[16];
};

__device__ __forceinline__ unsigned short f2bf(float f) {
  union { float f; unsigned u; } x; x.f = f;
  unsigned u = x.u;
  u += 0x7FFFu + ((u >> 16) & 1u);          // round-to-nearest-even
  return (unsigned short)(u >> 16);
}
__device__ __forceinline__ float bf2f(unsigned short h) {
  union { unsigned u; float f; } x; x.u = ((unsigned)h) << 16;
  return x.f;
}

#if defined(__has_builtin)
#if __has_builtin(__builtin_amdgcn_tensor_load_to_lds) && __has_builtin(__builtin_amdgcn_s_wait_tensorcnt)
#define USE_TDM 1
#endif
#endif

#ifdef USE_TDM
// TDM 2D tile load: `rows` rows of `rowlen` bf16 elements, source row pitch
// `stride` elements, packed contiguously into LDS at lds_byte.
// D# per CDNA5 ISA ch.8: group0 = {count, lds_addr, global_addr, type=2},
// group1 = {data_size=2B, tensor dims, tile dims, dim0 stride}. Groups 2/3
// zero (tile_dim2..4 = 0 -> unused); trailing v8i is the unused VADDR4 slot.
// This toolchain's builtin is the 6-arg (clang-23 / therock-10.0) form.
__device__ __forceinline__ void tdm_load_rows(unsigned lds_byte,
                                              const unsigned short* gptr,
                                              int rows, int rowlen, int stride) {
  unsigned long long ga = (unsigned long long)gptr;
  v4u g0;
  g0.x = 1u;                                                     // count = 1
  g0.y = (unsigned)__builtin_amdgcn_readfirstlane((int)lds_byte);
  g0.z = (unsigned)__builtin_amdgcn_readfirstlane((int)(unsigned)ga);
  g0.w = ((unsigned)__builtin_amdgcn_readfirstlane((int)(unsigned)(ga >> 32))) | (2u << 30);
  v8i g1;
  g1[0] = (1 << 16);                        // data_size = 1 (2 bytes/elem)
  g1[1] = (stride & 0xFFFF) << 16;          // tensor_dim0 lo16 (= full row pitch)
  g1[2] = ((stride >> 16) & 0xFFFF) | (rows << 16);  // tensor_dim0 hi | tensor_dim1 lo
  g1[3] = (rowlen & 0xFFFF) << 16;          // tile_dim0
  g1[4] = rows & 0xFFFF;                    // tile_dim1 (tile_dim2 = 0)
  g1[5] = stride;                           // tensor_dim0_stride lo32
  g1[6] = 0;
  g1[7] = 0;
  v4i gz = {0, 0, 0, 0};
  v8i gz8 = {0, 0, 0, 0, 0, 0, 0, 0};
  __builtin_amdgcn_tensor_load_to_lds(g0, g1, gz, gz, gz8, 0);
}
#endif

// Fold BN scale (g/sqrt(1+eps)) into the weights and convert to bf16.
// ws layout: [0, 32768) = w_kq' (128x256), [32768, 98304) = w_v' (256x256)
__global__ void prep_weights(const float* __restrict__ w_kq,
                             const float* __restrict__ g_kq,
                             const float* __restrict__ w_v,
                             const float* __restrict__ g_v,
                             unsigned short* __restrict__ wbf) {
  const float rs = rsqrtf(1.0f + EPS);
  int idx = blockIdx.x * blockDim.x + threadIdx.x;
  const int NKQ = HEADS * 2 * DKQ * IN_C;   // 32768
  const int NV  = HEADS * DV * IN_SKIP;     // 65536
  if (idx < NKQ) {
    int o = idx / IN_C;
    wbf[idx] = f2bf(w_kq[idx] * g_kq[o] * rs);
  } else if (idx < NKQ + NV) {
    int j = idx - NKQ;
    int o = j / IN_SKIP;
    wbf[idx] = f2bf(w_v[j] * g_v[o] * rs);
  }
}

// Stage a (256 x 64) fp32 matrix as bf16 transposed [d][c] into LDS
// (B-operand layout: N-major rows, K contiguous).
__device__ __forceinline__ void stage_T(const float* __restrict__ src,
                                        unsigned short* __restrict__ dst,
                                        int tid) {
  int c = tid;                              // 256 threads = 256 channels
  const float* p0 = src + (size_t)c * DIM;
  #pragma unroll
  for (int d0 = 0; d0 < DIM; d0 += 8) {
    float4 a0 = *reinterpret_cast<const float4*>(p0 + d0);
    float4 a1 = *reinterpret_cast<const float4*>(p0 + d0 + 4);
    float vals[8] = {a0.x, a0.y, a0.z, a0.w, a1.x, a1.y, a1.z, a1.w};
    #pragma unroll
    for (int k = 0; k < 8; ++k)
      dst[(d0 + k) * IN_C + c] = f2bf(vals[k]);
  }
}

__global__ __launch_bounds__(256)
void cross_axial_attn(const float* __restrict__ x_in,
                      const float* __restrict__ skip,
                      const unsigned short* __restrict__ wbf,
                      const float* __restrict__ b_kq,
                      const float* __restrict__ b_v,
                      const float* __restrict__ relative,
                      const float* __restrict__ g_attn,
                      const float* __restrict__ b_attn,
                      const float* __restrict__ g_out,
                      const float* __restrict__ b_out,
                      float* __restrict__ out) {
  // 32KB: phases 1-4: xT / skipT [64][256]; phase 5+: qk bf16 [128][64] (8192) + 8x2KB attn scratch
  __shared__ unsigned short sm1[16384];
  // 32KB: phases 1-2: per-wave 4KB TDM weight staging slots; phase 5+: v bf16 [h][i][j]
  __shared__ unsigned short smv[16384];
  // 12KB: relative bf16, rows padded to 128
  __shared__ unsigned short smrel[48 * 128];

  const int b    = blockIdx.x;
  const int tid  = threadIdx.x;
  const int w    = tid >> 5;        // wave id 0..7
  const int lane = tid & 31;
  const int l16  = lane & 15;
  const int hv   = lane >> 4;       // lane half (WMMA layout)
  const float rs = rsqrtf(1.0f + EPS);

#ifdef USE_TDM
  // Kick TDM load of this wave's W_kq rows (16 x 128 bf16, first K-half) into
  // its smv slot; overlaps with the xT staging below. TENSORcnt tracks it.
  const unsigned smv_byte = (unsigned)(unsigned long long)(const void*)smv + (unsigned)w * 4096u;
  tdm_load_rows(smv_byte, wbf + w * 16 * 256, 16, 128, 256);
#endif

  // ---- stage relative table (shared across all phases) ----
  for (int i = tid; i < 48 * 127; i += 256) {
    int c = i / 127, t = i - c * 127;
    smrel[c * 128 + t] = f2bf(relative[i]);
  }

  // ---- phase 1: xT into sm1 ----
  stage_T(x_in + (size_t)b * IN_C * DIM, sm1, tid);
  __syncthreads();

  // ---- phase 2: qk = W_kq' @ x  (128x64), keep D-frags in regs ----
  v8f qkacc[4];
  #pragma unroll
  for (int nt = 0; nt < 4; ++nt) qkacc[nt] = v8f{};
  {
#ifdef USE_TDM
    const unsigned short* smA = smv + w * 2048;   // this wave's staged 16x128 tile
    for (int kh = 0; kh < 2; ++kh) {
      __builtin_amdgcn_s_wait_tensorcnt((short)0);
      for (int k4 = 0; k4 < 4; ++k4) {
        int ks = kh * 4 + k4;
        BFrag A;
        int c0 = k4 * 32 + 16 * hv;
        A.q[0] = *reinterpret_cast<const uint4*>(&smA[l16 * 128 + c0]);
        A.q[1] = *reinterpret_cast<const uint4*>(&smA[l16 * 128 + c0 + 8]);
        #pragma unroll
        for (int nt = 0; nt < 4; ++nt) {
          BFrag Bf;
          int d  = nt * 16 + l16;
          int cb = ks * 32 + 16 * hv;
          Bf.q[0] = *reinterpret_cast<const uint4*>(&sm1[d * 256 + cb]);
          Bf.q[1] = *reinterpret_cast<const uint4*>(&sm1[d * 256 + cb + 8]);
          qkacc[nt] = __builtin_amdgcn_wmma_f32_16x16x32_bf16(
              false, A.v, false, Bf.v, (short)0, qkacc[nt], false, false);
        }
      }
      if (kh == 0) {
        // slot reuse: make sure our DS reads of the slot retired, then fetch K-half 1
        asm volatile("s_wait_dscnt 0x0" ::: "memory");
        tdm_load_rows(smv_byte, wbf + w * 16 * 256 + 128, 16, 128, 256);
      }
    }
#else
    const unsigned short* wq = wbf;               // [128][256]
    const int o = w * 16 + l16;                   // wave = M-tile
    for (int ks = 0; ks < 8; ++ks) {
      BFrag A;
      int c0 = ks * 32 + 16 * hv;
      A.q[0] = *reinterpret_cast<const uint4*>(wq + o * 256 + c0);
      A.q[1] = *reinterpret_cast<const uint4*>(wq + o * 256 + c0 + 8);
      #pragma unroll
      for (int nt = 0; nt < 4; ++nt) {
        BFrag Bf;
        int d  = nt * 16 + l16;
        int cb = ks * 32 + 16 * hv;
        Bf.q[0] = *reinterpret_cast<const uint4*>(&sm1[d * 256 + cb]);
        Bf.q[1] = *reinterpret_cast<const uint4*>(&sm1[d * 256 + cb + 8]);
        qkacc[nt] = __builtin_amdgcn_wmma_f32_16x16x32_bf16(
            false, A.v, false, Bf.v, (short)0, qkacc[nt], false, false);
      }
    }
#endif
  }
  __syncthreads();

  // ---- phase 3: skipT into sm1 ----
  stage_T(skip + (size_t)b * IN_SKIP * DIM, sm1, tid);
  __syncthreads();

  // ---- phase 4: v = W_v' @ skip  (256x64), A direct from L2-resident weights ----
  v8f vacc[2][4];
  #pragma unroll
  for (int m = 0; m < 2; ++m)
    #pragma unroll
    for (int nt = 0; nt < 4; ++nt) vacc[m][nt] = v8f{};
  {
    const unsigned short* wv = wbf + 32768;       // [256][256]
    #pragma unroll
    for (int mt2 = 0; mt2 < 2; ++mt2) {
      const int o = (2 * w + mt2) * 16 + l16;
      for (int ks = 0; ks < 8; ++ks) {
        BFrag A;
        int c0 = ks * 32 + 16 * hv;
        A.q[0] = *reinterpret_cast<const uint4*>(wv + o * 256 + c0);
        A.q[1] = *reinterpret_cast<const uint4*>(wv + o * 256 + c0 + 8);
        if (ks < 7) __builtin_prefetch(wv + o * 256 + (ks + 1) * 32, 0, 0);
        #pragma unroll
        for (int nt = 0; nt < 4; ++nt) {
          BFrag Bf;
          int d  = nt * 16 + l16;
          int cb = ks * 32 + 16 * hv;
          Bf.q[0] = *reinterpret_cast<const uint4*>(&sm1[d * 256 + cb]);
          Bf.q[1] = *reinterpret_cast<const uint4*>(&sm1[d * 256 + cb + 8]);
          vacc[mt2][nt] = __builtin_amdgcn_wmma_f32_16x16x32_bf16(
              false, A.v, false, Bf.v, (short)0, vacc[mt2][nt], false, false);
        }
      }
    }
  }
  __syncthreads();  // everyone done reading skipT; sm1/smv now free

  // ---- phase 5: commit qk (bf16, [o][d]) into sm1[0:8192], v (bf16, [h][i][j]) into smv ----
  {
    float bq[8];
    #pragma unroll
    for (int r = 0; r < 8; ++r) bq[r] = b_kq[w * 16 + r + 8 * hv];
    #pragma unroll
    for (int nt = 0; nt < 4; ++nt)
      #pragma unroll
      for (int r = 0; r < 8; ++r) {
        int o = w * 16 + r + 8 * hv;
        int d = nt * 16 + l16;
        sm1[o * 64 + d] = f2bf(qkacc[nt][r] + bq[r]);
      }
    #pragma unroll
    for (int mt2 = 0; mt2 < 2; ++mt2)
      #pragma unroll
      for (int r = 0; r < 8; ++r) {
        int o = (2 * w + mt2) * 16 + r + 8 * hv;
        float bias = b_v[o];
        int h2 = o & 7, iv = o >> 3;          // v[b,h,i,d] = v_full[b, i*8+h, d]
        #pragma unroll
        for (int nt = 0; nt < 4; ++nt) {
          int d = nt * 16 + l16;
          smv[(h2 * 32 + iv) * 64 + d] = f2bf(vacc[mt2][nt][r] + bias);
        }
      }
  }
  __syncthreads();

  // ---- phase 6: per-head attention; wave w owns head h = w ----
  const int h = w;
  unsigned short* qkbf = sm1;                       // [128][64]
  unsigned short* scr  = sm1 + 8192 + w * 1024;     // per-wave [16][64] attn scratch

  const float s0 = g_attn[h * 3 + 0] * rs;
  const float s1 = g_attn[h * 3 + 1] * rs;
  const float s2 = g_attn[h * 3 + 2] * rs;
  const float bsum = b_attn[h * 3 + 0] + b_attn[h * 3 + 1] + b_attn[h * 3 + 2];

  float so1[2], bo1[2], so2[2], bo2[2];
  #pragma unroll
  for (int it = 0; it < 2; ++it) {
    int c1 = h * 32 + it * 16 + l16;
    so1[it] = g_out[c1] * rs;       bo1[it] = b_out[c1];
    so2[it] = g_out[256 + c1] * rs; bo2[it] = b_out[256 + c1];
  }

  for (int dq = 0; dq < 4; ++dq) {                  // 16-row d-quarters
    // dots = q^T k : A (M=d, K=i pad 8->32), one WMMA per (dq, jt)
    BFrag Aq;
    #pragma unroll
    for (int t = 0; t < 16; ++t) Aq.s[t] = 0;
    {
      int d = dq * 16 + l16;
      #pragma unroll
      for (int t = 0; t < 8; ++t) {                 // K = t + 16*hv ; real i only at hv==0
        unsigned short qv = qkbf[(t * 8 + h) * 64 + d];
        Aq.s[t] = hv ? (unsigned short)0 : qv;
      }
    }
    v8f facc[4];
    #pragma unroll
    for (int jt = 0; jt < 4; ++jt) {
      BFrag Bk;
      #pragma unroll
      for (int t = 0; t < 16; ++t) Bk.s[t] = 0;
      int j = jt * 16 + l16;
      #pragma unroll
      for (int t = 0; t < 8; ++t) {                 // K = t + 16*hv ; real i only at hv==0
        unsigned short kk = qkbf[((t + 8) * 8 + h) * 64 + j];
        Bk.s[t] = hv ? (unsigned short)0 : kk;
      }
      v8f z = {};
      facc[jt] = __builtin_amdgcn_wmma_f32_16x16x32_bf16(
          false, Aq.v, false, Bk.v, (short)0, z, false, false);
    }

    // axial qr / kr (VALU) + BN affine + sum
    #pragma unroll
    for (int r = 0; r < 8; ++r) {
      int d = dq * 16 + r + 8 * hv;
      float qv[8], kv8[8];
      #pragma unroll
      for (int i = 0; i < 8; ++i) {
        qv[i]  = bf2f(qkbf[(i * 8 + h) * 64 + d]);
        kv8[i] = bf2f(qkbf[((i + 8) * 8 + h) * 64 + d]);
      }
      #pragma unroll
      for (int jt = 0; jt < 4; ++jt) {
        int j = jt * 16 + l16;
        int idx = d - j + 63;
        float qr = 0.f, kr = 0.f;
        #pragma unroll
        for (int i = 0; i < 8; ++i) {
          qr += qv[i]  * bf2f(smrel[i * 128 + idx]);
          kr += kv8[i] * bf2f(smrel[(8 + i) * 128 + idx]);
        }
        facc[jt][r] = facc[jt][r] * s2 + qr * s0 + kr * s1 + bsum;
      }
    }

    // softmax over j (rows live in 16-lane halves; shfl masks 1..8 stay inside)
    #pragma unroll
    for (int r = 0; r < 8; ++r) {
      float m = facc[0][r];
      #pragma unroll
      for (int jt = 1; jt < 4; ++jt) m = fmaxf(m, facc[jt][r]);
      for (int off = 1; off < 16; off <<= 1) m = fmaxf(m, __shfl_xor(m, off, 32));
      float sum = 0.f;
      #pragma unroll
      for (int jt = 0; jt < 4; ++jt) {
        float e = __expf(facc[jt][r] - m);
        facc[jt][r] = e; sum += e;
      }
      for (int off = 1; off < 16; off <<= 1) sum += __shfl_xor(sum, off, 32);
      float inv = 1.0f / sum;
      #pragma unroll
      for (int jt = 0; jt < 4; ++jt) facc[jt][r] *= inv;
    }

    // stash attn (bf16) in per-wave scratch (same-wave LDS ops are in-order)
    #pragma unroll
    for (int jt = 0; jt < 4; ++jt)
      #pragma unroll
      for (int r = 0; r < 8; ++r) {
        int dl = r + 8 * hv;
        int j  = jt * 16 + l16;
        scr[dl * 64 + j] = f2bf(facc[jt][r]);
      }

    // out = attn @ v^T via WMMA; A from scratch, B from smv
    BFrag Aat[2];
    #pragma unroll
    for (int kst = 0; kst < 2; ++kst) {
      int c0 = kst * 32 + 16 * hv;
      Aat[kst].q[0] = *reinterpret_cast<const uint4*>(&scr[l16 * 64 + c0]);
      Aat[kst].q[1] = *reinterpret_cast<const uint4*>(&scr[l16 * 64 + c0 + 8]);
    }
    v8f oacc2[2];
    #pragma unroll
    for (int it = 0; it < 2; ++it) {
      v8f oacc = {};
      int iidx = it * 16 + l16;
      #pragma unroll
      for (int kst = 0; kst < 2; ++kst) {
        BFrag Bv;
        int k0 = kst * 32 + 16 * hv;
        Bv.q[0] = *reinterpret_cast<const uint4*>(&smv[(h * 32 + iidx) * 64 + k0]);
        Bv.q[1] = *reinterpret_cast<const uint4*>(&smv[(h * 32 + iidx) * 64 + k0 + 8]);
        oacc = __builtin_amdgcn_wmma_f32_16x16x32_bf16(
            false, Aat[kst].v, false, Bv.v, (short)0, oacc, false, false);
      }
      oacc2[it] = oacc;
    }

    // axial kv (VALU): both i half-tiles share one attn read per (j, r)
    float kv2[2][8];
    #pragma unroll
    for (int r = 0; r < 8; ++r) { kv2[0][r] = 0.f; kv2[1][r] = 0.f; }
    for (int j = 0; j < 64; ++j) {
      #pragma unroll
      for (int r = 0; r < 8; ++r) {
        int dl = r + 8 * hv;
        int d  = dq * 16 + dl;
        int idx = d - j + 63;
        float a = bf2f(scr[dl * 64 + j]);
        kv2[0][r] += a * bf2f(smrel[(16 + l16) * 128 + idx]);
        kv2[1][r] += a * bf2f(smrel[(32 + l16) * 128 + idx]);
      }
    }

    // final affine + sum of [kv, out] stack, write output
    #pragma unroll
    for (int it = 0; it < 2; ++it) {
      int iidx = it * 16 + l16;
      #pragma unroll
      for (int r = 0; r < 8; ++r) {
        int d = dq * 16 + r + 8 * hv;
        float res = (kv2[it][r] * so1[it] + bo1[it]) + (oacc2[it][r] * so2[it] + bo2[it]);
        out[(size_t)b * (256 * DIM) + (h * 32 + iidx) * 64 + d] = res;
      }
    }
  }
}

extern "C" void kernel_launch(void* const* d_in, const int* in_sizes, int n_in,
                              void* d_out, int out_size, void* d_ws, size_t ws_size,
                              hipStream_t stream) {
  const float* x_in     = (const float*)d_in[0];
  const float* skip     = (const float*)d_in[1];
  const float* w_kq     = (const float*)d_in[2];
  const float* g_kq     = (const float*)d_in[3];
  const float* b_kq     = (const float*)d_in[4];
  const float* w_v      = (const float*)d_in[5];
  const float* g_v      = (const float*)d_in[6];
  const float* b_v      = (const float*)d_in[7];
  const float* relative = (const float*)d_in[8];
  const float* g_attn   = (const float*)d_in[9];
  const float* b_attn   = (const float*)d_in[10];
  const float* g_out    = (const float*)d_in[11];
  const float* b_out    = (const float*)d_in[12];
  unsigned short* wbf   = (unsigned short*)d_ws;

  const int B = in_sizes[0] / (IN_C * DIM);
  const int NW = HEADS * 2 * DKQ * IN_C + HEADS * DV * IN_SKIP;  // 98304

  prep_weights<<<(NW + 255) / 256, 256, 0, stream>>>(w_kq, g_kq, w_v, g_v, wbf);
  cross_axial_attn<<<B, 256, 0, stream>>>(x_in, skip, wbf, b_kq, b_v, relative,
                                          g_attn, b_attn, g_out, b_out,
                                          (float*)d_out);
}
